// FastMQAttention_8658654068813
// MI455X (gfx1250) — matrix operand order repaired and checked by reading, not compiled, not run on hardware
//
#include <hip/hip_runtime.h>

#define HEAD_DIM 128
#define SEQ      2048
#define NHEADS   32
#define NBATCH   4
#define KBLK     32     // keys per iteration
#define QTILE    16     // q rows per wave
#define WAVES    8      // waves per block -> 128 q rows per block

typedef __attribute__((ext_vector_type(16))) __bf16 v16bf;
typedef __attribute__((ext_vector_type(8)))  __bf16 v8bf;
typedef __attribute__((ext_vector_type(8)))  float  v8f;

__device__ __forceinline__ v16bf join8(v8bf lo, v8bf hi) {
  return __builtin_shufflevector(lo, hi, 0,1,2,3,4,5,6,7,8,9,10,11,12,13,14,15);
}

__global__ __launch_bounds__(256)
void mqa_flash_bf16_wmma(const float* __restrict__ Qg,
                         const float* __restrict__ Kg,
                         const float* __restrict__ Vg,
                         float* __restrict__ Og) {
  // Double-buffered LDS staging:
  //   sK : K block row-major bf16 [key][d]
  //   sVT: V block transposed bf16 [d][key]
  __shared__ __bf16 sK [2][KBLK * HEAD_DIM];   // 2 x 8 KB
  __shared__ __bf16 sVT[2][HEAD_DIM * KBLK];   // 2 x 8 KB

  const int tid   = threadIdx.x;
  const int lane  = tid & 31;
  const int wave  = tid >> 5;
  const int lq    = lane & 15;   // q-column (B/C layouts) or key-row (A layout)
  const int lhalf = lane >> 4;   // which 16-lane half

  const int blk  = blockIdx.x;
  const int qgrp = blk & 15;           // 16 groups of 128 q-rows per head
  const int h    = (blk >> 4) & 31;
  const int b    = blk >> 9;

  const int q0 = qgrp * (WAVES * QTILE) + wave * QTILE;

  const float scale = 0.08838834764831845f; // 1/sqrt(128), folded into Q

  // ---------------- Q fragments (B-matrix layout for S^T = K * Q^T) --------
  // lane holds q-col = lq ; elements 0-7 hold d = c*32 + 8*lhalf + e ; elements 8-15 -> +16
  const float* qrow = Qg + (((size_t)b * NHEADS + h) * SEQ + (q0 + lq)) * HEAD_DIM;
  v16bf qf[4];
#pragma unroll
  for (int c = 0; c < 4; ++c) {
    const float4* p = (const float4*)(qrow + c * 32 + 8 * lhalf);
    float tf[16];
    *(float4*)&tf[0]  = p[0];
    *(float4*)&tf[4]  = p[1];
    *(float4*)&tf[8]  = p[4];
    *(float4*)&tf[12] = p[5];
    v16bf q;
#pragma unroll
    for (int e = 0; e < 16; ++e) q[e] = (__bf16)(tf[e] * scale);
    qf[c] = q;
  }

  const v8f vzero = {0.f,0.f,0.f,0.f,0.f,0.f,0.f,0.f};
  v8f o[8];
#pragma unroll
  for (int t = 0; t < 8; ++t) o[t] = vzero;

  float mrow = -3.0e38f;   // running row max (per q = lq, duplicated on lane halves)
  float lrow = 0.0f;       // running row sum

  const float* Kb = Kg + (size_t)b * SEQ * HEAD_DIM;
  const float* Vb = Vg + (size_t)b * SEQ * HEAD_DIM;

  // cooperative staging indices: 8 threads per key row, 4 float4 each
  const int ckey = tid >> 3;   // 0..31
  const int cf4  = tid & 7;    // 0..7

  float4 kv[4], vv[4];   // in-flight staged block

  auto load_blk = [&](int kb) {
    const float4* kr = (const float4*)(Kb + (size_t)(kb + ckey) * HEAD_DIM);
    const float4* vr = (const float4*)(Vb + (size_t)(kb + ckey) * HEAD_DIM);
#pragma unroll
    for (int j = 0; j < 4; ++j) {
      kv[j] = kr[cf4 + j * 8];
      vv[j] = vr[cf4 + j * 8];
    }
  };
  auto store_blk = [&](int buf) {
#pragma unroll
    for (int j = 0; j < 4; ++j) {
      const int f4i = cf4 + j * 8;      // 0..31
      const int d0  = f4i * 4;
      __bf16* kd = &sK[buf][ckey * HEAD_DIM + d0];
      kd[0] = (__bf16)kv[j].x; kd[1] = (__bf16)kv[j].y;
      kd[2] = (__bf16)kv[j].z; kd[3] = (__bf16)kv[j].w;
      sVT[buf][(d0 + 0) * KBLK + ckey] = (__bf16)vv[j].x;
      sVT[buf][(d0 + 1) * KBLK + ckey] = (__bf16)vv[j].y;
      sVT[buf][(d0 + 2) * KBLK + ckey] = (__bf16)vv[j].z;
      sVT[buf][(d0 + 3) * KBLK + ckey] = (__bf16)vv[j].w;
    }
  };

  // prologue: stage block 0
  load_blk(0);
  store_blk(0);
  __syncthreads();

  int cur = 0;
  for (int kb = 0; kb < SEQ; kb += KBLK) {
    const int nxt = kb + KBLK;
    // issue next block's global loads before compute (overlap VMEM with WMMA)
    if (nxt < SEQ) {
      load_blk(nxt);
      if (nxt + KBLK < SEQ) {   // keep L2 warm two blocks ahead
        __builtin_prefetch(Kb + (size_t)(nxt + KBLK + ckey) * HEAD_DIM + cf4 * 16, 0, 3);
        __builtin_prefetch(Vb + (size_t)(nxt + KBLK + ckey) * HEAD_DIM + cf4 * 16, 0, 3);
      }
    }

    const __bf16* sKc = sK[cur];
    const __bf16* sVc = sVT[cur];

    // ---------------- S^T tiles: st[sub] = K[sub*16..][*] x Q^T ------------
    v8f st0 = vzero, st1 = vzero;
#pragma unroll
    for (int c = 0; c < 4; ++c) {
      // A-fragment of K: lane holds key-row lq (sub 0) / 16+lq (sub 1),
      // elements 0-7 -> d = c*32 + 8*lhalf + e ; elements 8-15 -> +16
      const __bf16* k0 = &sKc[lq * HEAD_DIM + c * 32 + 8 * lhalf];
      const __bf16* k1 = &sKc[(16 + lq) * HEAD_DIM + c * 32 + 8 * lhalf];
      v16bf a0 = join8(*(const v8bf*)k0, *(const v8bf*)(k0 + 16));
      v16bf a1 = join8(*(const v8bf*)k1, *(const v8bf*)(k1 + 16));
      st0 = __builtin_amdgcn_wmma_f32_16x16x32_bf16(false, a0, false, qf[c],
                                                    (short)0, st0, false, false);
      st1 = __builtin_amdgcn_wmma_f32_16x16x32_bf16(false, a1, false, qf[c],
                                                    (short)0, st1, false, false);
    }

    // ---------------- online softmax (fp32) --------------------------------
    float tmax = st0[0];
#pragma unroll
    for (int r = 1; r < 8; ++r) tmax = fmaxf(tmax, st0[r]);
#pragma unroll
    for (int r = 0; r < 8; ++r) tmax = fmaxf(tmax, st1[r]);
    tmax = fmaxf(tmax, __shfl_xor(tmax, 16, 32));   // combine key halves

    const float mnew  = fmaxf(mrow, tmax);
    const float alpha = __expf(mrow - mnew);

    float psum = 0.f;
    v16bf pf;   // P in A-layout: els 0-7 keys 8*lhalf+r, els 8-15 keys 16+8*lhalf+r
#pragma unroll
    for (int r = 0; r < 8; ++r) {
      float p0 = __expf(st0[r] - mnew);
      float p1 = __expf(st1[r] - mnew);
      psum += p0 + p1;
      pf[r]     = (__bf16)p0;
      pf[8 + r] = (__bf16)p1;
    }
    psum += __shfl_xor(psum, 16, 32);
    lrow = lrow * alpha + psum;
    mrow = mnew;

    // rescale O accumulators: q row of o[t][r] is r + 8*lhalf -> broadcast alpha
#pragma unroll
    for (int r = 0; r < 8; ++r) {
      const float ar = __shfl(alpha, 8 * lhalf + r, 32);
#pragma unroll
      for (int t = 0; t < 8; ++t) o[t][r] *= ar;
    }

    // ---------------- O += P x V -------------------------------------------
#pragma unroll
    for (int t = 0; t < 8; ++t) {
      // B-fragment of V: lane holds d-col t*16+lq, els 0-7 keys 8*lhalf+r, els 8-15 keys 16+8*lhalf+r
      const __bf16* vp = &sVc[(t * 16 + lq) * KBLK + 8 * lhalf];
      v16bf bv = join8(*(const v8bf*)vp, *(const v8bf*)(vp + 16));
      o[t] = __builtin_amdgcn_wmma_f32_16x16x32_bf16(false, pf, false, bv,
                                                     (short)0, o[t], false, false);
    }

    // convert + store next block into the other buffer, then single barrier
    if (nxt < SEQ) store_blk(cur ^ 1);
    __syncthreads();
    cur ^= 1;
  }

  // ---------------- normalize and store -----------------------------------
  const float linv = 1.0f / lrow;
  float* obase = Og + (((size_t)b * NHEADS + h) * SEQ + q0) * HEAD_DIM;
#pragma unroll
  for (int r = 0; r < 8; ++r) {
    const float lr  = __shfl(linv, 8 * lhalf + r, 32);
    const int   qrw = r + 8 * lhalf;
    float* dst = obase + (size_t)qrw * HEAD_DIM;
#pragma unroll
    for (int t = 0; t < 8; ++t) dst[t * 16 + lq] = o[t][r] * lr;
  }
}

extern "C" void kernel_launch(void* const* d_in, const int* in_sizes, int n_in,
                              void* d_out, int out_size, void* d_ws, size_t ws_size,
                              hipStream_t stream) {
  const float* Q = (const float*)d_in[0];
  const float* K = (const float*)d_in[1];
  const float* V = (const float*)d_in[2];
  float* O = (float*)d_out;
  (void)in_sizes; (void)n_in; (void)out_size; (void)d_ws; (void)ws_size;

  dim3 grid(NBATCH * NHEADS * (SEQ / (WAVES * QTILE)));  // 4*32*16 = 2048
  dim3 block(WAVES * 32);                                // 256 threads, 8 waves
  mqa_flash_bf16_wmma<<<grid, block, 0, stream>>>(Q, K, V, O);
}
